// DoubleNet_45552423141937
// MI455X (gfx1250) — compile-verified, matrix operand order checked by reading, not run on hardware
//
#include <hip/hip_runtime.h>

typedef __attribute__((ext_vector_type(16))) _Float16 v16h;
typedef __attribute__((ext_vector_type(8)))  float    v8f;

#define L2E  1.4426950408889634f
#define LN2  0.6931471805599453f
#define LOG5 1.6094379124341003f

// Packed weight layout offsets (in _Float16 elements). One fragment = 512 halfs
// (32 lanes x 16 halfs), fragment id = nt*ktiles + kt, lane-major inside.
#define OFF_NN1 0       // 25->128 : ktiles=1, ntiles=8  -> 4096
#define OFF_NN2 4096    // 128->128: ktiles=4, ntiles=8  -> 16384
#define OFF_NN3 20480   // 128->128
#define OFF_NN4 36864   // 128->25 : ktiles=4, ntiles=2  -> 4096
#define OFF_P1  40960
#define OFF_P2  45056
#define OFF_P3  61440
#define OFF_P4  77824   // 128->5  : ktiles=4, ntiles=1  -> 2048
#define PACK_TOTAL 79872

// ---------------------------------------------------------------------------
// Prepack: fp32 weights (stored [out][in]) -> f16 WMMA B-fragment layout.
// 16-bit B operand (KxN = 32x16), mirrored from the ISA A-matrix 16x32 table:
//   lanes 0-15: n=lane,   halfs h<8 -> k=h,      h>=8 -> k=h+8
//   lanes16-31: n=lane-16, halfs h<8 -> k=h+8,   h>=8 -> k=h+16
// ---------------------------------------------------------------------------
__global__ void pack_w_kernel(const float* __restrict__ W, _Float16* __restrict__ dst,
                              int Kr, int Nr, int ktiles, int ntiles)
{
    int idx = blockIdx.x * blockDim.x + threadIdx.x;
    int total = ktiles * ntiles * 512;
    if (idx >= total) return;
    int h    = idx & 15;
    int lane = (idx >> 4) & 31;
    int frag = idx >> 9;
    int kt = frag % ktiles;
    int nt = frag / ktiles;
    int base = (lane >= 16) ? 8 : 0;
    int kl = (h < 8) ? (h + base) : (h + 8 + base);
    int n = nt * 16 + (lane & 15);
    int k = kt * 32 + kl;
    float v = (n < Nr && k < Kr) ? W[n * Kr + k] : 0.0f;
    dst[idx] = (_Float16)v;
}

// A fragment (16x32 f16) for lane from an LDS activation row (stride 128 halfs).
// p points at act[m] + kt*32 + base; needed dwords are {0,1,2,3, 8,9,10,11}.
__device__ __forceinline__ v16h load_afrag_lds(const _Float16* p)
{
    const unsigned int* q = (const unsigned int*)p;
    union { unsigned int u[8]; v16h v; } a;
    a.u[0] = q[0]; a.u[1] = q[1]; a.u[2] = q[2];  a.u[3] = q[3];
    a.u[4] = q[8]; a.u[5] = q[9]; a.u[6] = q[10]; a.u[7] = q[11];
    return a.v;
}

// A fragment from a padded fp32 row (layer 1). p = xs[row] + base.
__device__ __forceinline__ v16h make_afrag_f32(const float* p)
{
    v16h a;
#pragma unroll
    for (int h = 0; h < 8; ++h)  a[h] = (_Float16)p[h];
#pragma unroll
    for (int h = 8; h < 16; ++h) a[h] = (_Float16)p[h + 8];
    return a;
}

__device__ __forceinline__ float tanh_fast(float x)
{
    // tanh(x) = 1 - 2/(exp(2x)+1); saturates cleanly at +-1 (rcp(inf)=0)
    float e = __builtin_amdgcn_exp2f(x * (2.0f * L2E));
    return 1.0f - 2.0f * __builtin_amdgcn_rcpf(e + 1.0f);
}

// One 128-wide hidden layer: acc = A @ B, +bias, tanh, scatter to act tile.
// D layout: element (m = r + (lane>=16?8:0), n = nt*16 + (lane&15)).
// nt loop is deliberately NOT unrolled to keep VGPR pressure low (one B
// fragment + one accumulator live at a time alongside the a[] fragments).
template<int KT>
__device__ __forceinline__ void hidden_layer(const v16h* a,
                                             const _Float16* __restrict__ wl,
                                             const float* __restrict__ bias,
                                             _Float16* actdst, int lane)
{
    int l15 = lane & 15;
    int mb  = (lane >= 16) ? 8 : 0;
#pragma unroll 1
    for (int nt = 0; nt < 8; ++nt) {
        v8f acc = {};
#pragma unroll
        for (int kt = 0; kt < KT; ++kt) {
            v16h b = *(const v16h*)(wl + (nt * KT + kt) * 512 + lane * 16);
            acc = __builtin_amdgcn_wmma_f32_16x16x32_f16(
                false, a[kt], false, b, (short)0, acc, false, false);
        }
        int n = nt * 16 + l15;
        float bn = bias[n];
#pragma unroll
        for (int r = 0; r < 8; ++r)
            actdst[(r + mb) * 128 + n] = (_Float16)tanh_fast(acc[r] + bn);
    }
}

// ---------------------------------------------------------------------------
// Fused kernel: 128 threads = 4 waves, 128 samples/block (2 M-tiles per wave)
// __launch_bounds__(128, 4): cap VGPRs at <=256 (no vgpr-msb, no spills),
// target >=4 waves/SIMD occupancy.
// ---------------------------------------------------------------------------
__global__ __launch_bounds__(128, 4)
void doublenet_fused(const float* __restrict__ bids, const _Float16* __restrict__ pk,
                     const float* __restrict__ nb1, const float* __restrict__ nb2,
                     const float* __restrict__ nb3, const float* __restrict__ nb4,
                     const float* __restrict__ pb1, const float* __restrict__ pb2,
                     const float* __restrict__ pb3, const float* __restrict__ pb4,
                     float* __restrict__ out_alloc, float* __restrict__ out_pay,
                     int nbatch)
{
    __shared__ float    xs[128][32];          // X padded K=32 (fp32)  16 KB
    __shared__ _Float16 act[4][2][16][128];   // per-wave ping/pong    32 KB
    __shared__ float    cl[128][25];          // clamped bids          12.5 KB
    __shared__ float    fr[128][5];           // payment fractions      2.5 KB

    const int tid = threadIdx.x;
    const int blockBase = blockIdx.x * 128;

    // Stage X (zero-padded to 32 columns)
#pragma unroll 1
    for (int i = tid; i < 128 * 32; i += 128) {
        int r = i >> 5, c = i & 31;
        float v = 0.0f;
        if (c < 25 && (blockBase + r) < nbatch)
            v = bids[(size_t)(blockBase + r) * 25 + c];
        xs[r][c] = v;
    }
    __syncthreads();

    const int w    = tid >> 5;
    const int lane = tid & 31;
    const int l15  = lane & 15;
    const int mb   = (lane >= 16) ? 8 : 0;   // row offset of this half-wave in D
    const int kb   = mb;                      // K offset of this half-wave in A

    _Float16* act0 = &act[w][0][0][0];
    _Float16* act1 = &act[w][1][0][0];

#pragma unroll 1
    for (int mt = 0; mt < 2; ++mt) {
        const int row0 = w * 32 + mt * 16;
        v16h a[4];

        // ------------------ allocation network ------------------
        a[0] = make_afrag_f32(&xs[row0 + l15][kb]);
        hidden_layer<1>(a, pk + OFF_NN1, nb1, act0, lane);
#pragma unroll
        for (int kt = 0; kt < 4; ++kt) a[kt] = load_afrag_lds(&act0[l15 * 128 + kt * 32 + kb]);
        hidden_layer<4>(a, pk + OFF_NN2, nb2, act1, lane);
#pragma unroll
        for (int kt = 0; kt < 4; ++kt) a[kt] = load_afrag_lds(&act1[l15 * 128 + kt * 32 + kb]);
        hidden_layer<4>(a, pk + OFF_NN3, nb3, act0, lane);
#pragma unroll
        for (int kt = 0; kt < 4; ++kt) a[kt] = load_afrag_lds(&act0[l15 * 128 + kt * 32 + kb]);
        // final linear 128->25, clamp by X
#pragma unroll 1
        for (int nt = 0; nt < 2; ++nt) {
            v8f acc = {};
#pragma unroll
            for (int kt = 0; kt < 4; ++kt) {
                v16h b = *(const v16h*)(pk + OFF_NN4 + (nt * 4 + kt) * 512 + lane * 16);
                acc = __builtin_amdgcn_wmma_f32_16x16x32_f16(
                    false, a[kt], false, b, (short)0, acc, false, false);
            }
            int n = nt * 16 + l15;
            if (n < 25) {
                float bn = nb4[n];
#pragma unroll
                for (int r = 0; r < 8; ++r) {
                    int m = row0 + r + mb;
                    cl[m][n] = fminf(xs[m][n], acc[r] + bn);
                }
            }
        }

        // ------------------ payment network ------------------
        a[0] = make_afrag_f32(&xs[row0 + l15][kb]);
        hidden_layer<1>(a, pk + OFF_P1, pb1, act0, lane);
#pragma unroll
        for (int kt = 0; kt < 4; ++kt) a[kt] = load_afrag_lds(&act0[l15 * 128 + kt * 32 + kb]);
        hidden_layer<4>(a, pk + OFF_P2, pb2, act1, lane);
#pragma unroll
        for (int kt = 0; kt < 4; ++kt) a[kt] = load_afrag_lds(&act1[l15 * 128 + kt * 32 + kb]);
        hidden_layer<4>(a, pk + OFF_P3, pb3, act0, lane);
#pragma unroll
        for (int kt = 0; kt < 4; ++kt) a[kt] = load_afrag_lds(&act0[l15 * 128 + kt * 32 + kb]);
        {   // final linear 128->5, sigmoid
            v8f acc = {};
#pragma unroll
            for (int kt = 0; kt < 4; ++kt) {
                v16h b = *(const v16h*)(pk + OFF_P4 + kt * 512 + lane * 16);
                acc = __builtin_amdgcn_wmma_f32_16x16x32_f16(
                    false, a[kt], false, b, (short)0, acc, false, false);
            }
            int n = l15;
            if (n < 5) {
                float bn = pb4[n];
#pragma unroll
                for (int r = 0; r < 8; ++r) {
                    int m = row0 + r + mb;
                    float x = acc[r] + bn;
                    fr[m][n] = __builtin_amdgcn_rcpf(1.0f + __builtin_amdgcn_exp2f(-x * L2E));
                }
            }
        }
    }
    __syncthreads();

    // ------------------ per-sample 6x6 log-Sinkhorn ------------------
    {
        const int r = tid;                       // sample-in-block
        const size_t s = (size_t)blockBase + r;
        float lp[6][6];
#pragma unroll
        for (int i = 0; i < 6; ++i)
#pragma unroll
            for (int j = 0; j < 6; ++j)
                lp[i][j] = (i < 5 && j < 5) ? cl[r][i * 5 + j] * 10.0f : 0.0f; // /EPS

#pragma unroll 1
        for (int it = 0; it < 20; ++it) {
            // row normalization (axis=2), log_a = [0,0,0,0,0,log5]
#pragma unroll
            for (int i = 0; i < 6; ++i) {
                float m = lp[i][0];
#pragma unroll
                for (int j = 1; j < 6; ++j) m = fmaxf(m, lp[i][j]);
                float se = 0.0f;
#pragma unroll
                for (int j = 0; j < 6; ++j) se += __builtin_amdgcn_exp2f((lp[i][j] - m) * L2E);
                float adj = m + __builtin_amdgcn_logf(se) * LN2 - ((i == 5) ? LOG5 : 0.0f);
#pragma unroll
                for (int j = 0; j < 6; ++j) lp[i][j] -= adj;
            }
            // column normalization (axis=1), log_b = [0,0,0,0,0,log5]
#pragma unroll
            for (int j = 0; j < 6; ++j) {
                float m = lp[0][j];
#pragma unroll
                for (int i = 1; i < 6; ++i) m = fmaxf(m, lp[i][j]);
                float se = 0.0f;
#pragma unroll
                for (int i = 0; i < 6; ++i) se += __builtin_amdgcn_exp2f((lp[i][j] - m) * L2E);
                float adj = m + __builtin_amdgcn_logf(se) * LN2 - ((j == 5) ? LOG5 : 0.0f);
#pragma unroll
                for (int i = 0; i < 6; ++i) lp[i][j] -= adj;
            }
        }

        if (s < (size_t)nbatch) {
#pragma unroll
            for (int a5 = 0; a5 < 5; ++a5) {
                float dot = 0.0f;
#pragma unroll
                for (int j = 0; j < 5; ++j) {
                    float al = __builtin_amdgcn_exp2f(lp[a5][j] * L2E);
                    out_alloc[s * 25 + a5 * 5 + j] = al;
                    dot += al * xs[r][a5 * 5 + j];
                }
                out_pay[s * 5 + a5] = fr[r][a5] * dot;
            }
        }
    }
}

extern "C" void kernel_launch(void* const* d_in, const int* in_sizes, int n_in,
                              void* d_out, int out_size, void* d_ws, size_t ws_size,
                              hipStream_t stream)
{
    const float* bids = (const float*)d_in[0];
    const float* nnW1 = (const float*)d_in[1];
    const float* nb1  = (const float*)d_in[2];
    const float* nnW2 = (const float*)d_in[3];
    const float* nb2  = (const float*)d_in[4];
    const float* nnW3 = (const float*)d_in[5];
    const float* nb3  = (const float*)d_in[6];
    const float* nnW4 = (const float*)d_in[7];
    const float* nb4  = (const float*)d_in[8];
    const float* pW1  = (const float*)d_in[9];
    const float* pb1  = (const float*)d_in[10];
    const float* pW2  = (const float*)d_in[11];
    const float* pb2  = (const float*)d_in[12];
    const float* pW3  = (const float*)d_in[13];
    const float* pb3  = (const float*)d_in[14];
    const float* pW4  = (const float*)d_in[15];
    const float* pb4  = (const float*)d_in[16];

    const int nbatch = in_sizes[0] / 25;
    _Float16* pk = (_Float16*)d_ws;

    // Prepack all 8 weight matrices into WMMA B-fragment layout (fp32 -> f16)
    struct { const float* W; int off, Kr, Nr, kt, nt; } packs[8] = {
        { nnW1, OFF_NN1,  25, 128, 1, 8 },
        { nnW2, OFF_NN2, 128, 128, 4, 8 },
        { nnW3, OFF_NN3, 128, 128, 4, 8 },
        { nnW4, OFF_NN4, 128,  25, 4, 2 },
        { pW1,  OFF_P1,   25, 128, 1, 8 },
        { pW2,  OFF_P2,  128, 128, 4, 8 },
        { pW3,  OFF_P3,  128, 128, 4, 8 },
        { pW4,  OFF_P4,  128,   5, 4, 1 },
    };
    for (int i = 0; i < 8; ++i) {
        int total = packs[i].kt * packs[i].nt * 512;
        pack_w_kernel<<<(total + 255) / 256, 256, 0, stream>>>(
            packs[i].W, pk + packs[i].off, packs[i].Kr, packs[i].Nr,
            packs[i].kt, packs[i].nt);
    }

    float* out = (float*)d_out;
    int blocks = (nbatch + 127) / 128;
    doublenet_fused<<<blocks, 128, 0, stream>>>(
        bids, pk, nb1, nb2, nb3, nb4, pb1, pb2, pb3, pb4,
        out, out + (size_t)nbatch * 25, nbatch);
}
